// CPWanSelfAttention_2723009265848
// MI455X (gfx1250) — matrix operand order repaired and checked by reading, not compiled, not run on hardware
//
#include <hip/hip_runtime.h>
#include <hip/hip_bf16.h>
#include <stdint.h>

// Problem constants (from reference)
#define B_    2
#define S_    2048
#define DIM_  2048
#define H_    16
#define HD_   128
#define MROWS (B_ * S_)          // 4096 token rows

typedef __attribute__((ext_vector_type(16))) __bf16 v16bf;
typedef __attribute__((ext_vector_type(8)))  float  v8f;
typedef __attribute__((ext_vector_type(4)))  unsigned int v4u;
typedef __attribute__((ext_vector_type(8)))  int    v8i;
typedef __attribute__((ext_vector_type(4)))  int    v4i;

union Frag {                      // one WMMA A/B operand (16 bf16 = 8 VGPRs)
    v16bf  v;
    uint4  q[2];
    __bf16 e[16];
};
union Pack8 { __bf16 e[8]; uint4 u; };
union Pack4 { __bf16 e[4]; uint2 u; };

static __device__ __forceinline__ v8f zero8() {
    v8f z;
#pragma unroll
    for (int i = 0; i < 8; ++i) z[i] = 0.0f;
    return z;
}

// ---------------------------------------------------------------------------
// TDM: issue a 2D tensor_load_to_lds of a (rows x 32) bf16 tile whose rows are
// K-strided in global memory, padded in LDS to an 80B row stride (LSTR=40).
//   D# group0: count=1, lds_addr, 57b global addr, type=2
//   D# group1: data_size=2B, pad_enable, pad_interval=16dw, pad_amount=4dw,
//              tensor_dim0=k_len, tensor_dim1=huge, tile_dim0=32,
//              tile_dim1=rows, tensor_dim0_stride=k_len
// ---------------------------------------------------------------------------
static __device__ __forceinline__ void tdm_load_tile(const void* gsrc,
                                                     unsigned lds_addr,
                                                     int rows, int k_len) {
    unsigned long long ga = (unsigned long long)(uintptr_t)gsrc;
    v4u g0;
    g0[0] = 1u;                                            // count=1 (valid, user)
    g0[1] = lds_addr;                                      // LDS byte address
    g0[2] = (unsigned)(ga & 0xFFFFFFFFu);                  // global_addr[31:0]
    g0[3] = (unsigned)((ga >> 32) & 0x01FFFFFFu)           // global_addr[56:32]
          | 0x80000000u;                                   // type=2 ("image")

    v8i g1;
    g1[0] = (int)((1u << 16)        // data_size = 2 bytes
                | (1u << 20)        // pad_enable
                | (3u << 22)        // pad_interval: 16 dwords (one 64B row)
                | (3u << 25));      // pad_amount: 4 dwords (16B)  -> 80B stride
    g1[1] = (int)(((unsigned)k_len & 0xFFFFu) << 16);      // tensor_dim0[15:0]
    g1[2] = (int)((((unsigned)k_len >> 16) & 0xFFFFu)      // tensor_dim0[31:16]
                | 0xFFFF0000u);                            // tensor_dim1[15:0]
    g1[3] = (int)(0x7FFFu | (32u << 16));                  // dim1 hi | tile_dim0=32
    g1[4] = rows & 0xFFFF;                                 // tile_dim1, tile_dim2=0
    g1[5] = k_len;                                         // tensor_dim0_stride lo
    g1[6] = 0;                                             // stride hi | dim1_stride
    g1[7] = 0;

    v4i z4 = {0, 0, 0, 0};
#if defined(__clang_major__) && (__clang_major__ >= 23)
    v8i z8 = {0, 0, 0, 0, 0, 0, 0, 0};
    __builtin_amdgcn_tensor_load_to_lds(g0, g1, z4, z4, z8, 0);
#else
    __builtin_amdgcn_tensor_load_to_lds(g0, g1, z4, z4, 0);
#endif
}

// ---------------------------------------------------------------------------
// Kernel 1: f32 -> bf16 conversion (vectorized x4)
// ---------------------------------------------------------------------------
__global__ __launch_bounds__(256) void cvt_f32_bf16(const float* __restrict__ in,
                                                    __bf16* __restrict__ out, int n) {
    int i = (blockIdx.x * 256 + threadIdx.x) * 4;
    if (i + 3 < n) {
        float4 v = *(const float4*)(in + i);
        Pack4 p;
        p.e[0] = (__bf16)v.x; p.e[1] = (__bf16)v.y;
        p.e[2] = (__bf16)v.z; p.e[3] = (__bf16)v.w;
        *(uint2*)(out + i) = p.u;
    }
}

// ---------------------------------------------------------------------------
// Kernel 2: NT-GEMM  C[M,N] = A[M,K](bf16) * W[N,K]^T(bf16) + bias, C = f32
// Block tile 256x128, BK=32; 8 waves (4x2), each wave 64x64 via 4x4 WMMA frags.
// Tiles staged to LDS by the Tensor Data Mover, double-buffered: wave 0 issues
// the next K-tile's DMA, then waits TENSORcnt<=2 so only the current pair must
// be complete -> DMA overlaps all waves' WMMA work.
// ---------------------------------------------------------------------------
#define BM 256
#define BN 128
#define BK 32
#define LSTR 40   // padded LDS row stride in bf16 (32 + 8), set via TDM pad

__global__ __launch_bounds__(256, 1) void gemm_nt_bias(
    const __bf16* __restrict__ A,       // M x K
    const __bf16* __restrict__ Wall,    // z-stacked: N x K per z
    const float* __restrict__ b0, const float* __restrict__ b1, const float* __restrict__ b2,
    float* __restrict__ o0, float* __restrict__ o1, float* __restrict__ o2,
    int M, int N, int K)
{
    const int z = blockIdx.z;
    const __bf16* Bw   = Wall + (size_t)z * (size_t)N * (size_t)K;
    const float*  bias = (z == 0) ? b0 : ((z == 1) ? b1 : b2);
    float*        C    = (z == 0) ? o0 : ((z == 1) ? o1 : o2);

    __shared__ __bf16 As[2][BM * LSTR];   // 2 x 20KB
    __shared__ __bf16 Bs[2][BN * LSTR];   // 2 x 10KB

    const int t     = threadIdx.x;
    const int wave  = t >> 5;
    const int lane  = t & 31;
    const int lg    = lane >> 4;      // half-wave group
    const int lr    = lane & 15;
    const int waveM = wave >> 1;      // 0..3  -> 64-row slice
    const int waveN = wave & 1;       // 0..1  -> 64-col slice

    const int blockM = blockIdx.y * BM;
    const int blockN = blockIdx.x * BN;

    const __bf16* gA = A  + (size_t)blockM * K;
    const __bf16* gB = Bw + (size_t)blockN * K;
    const unsigned ldsA0 = (unsigned)(uintptr_t)&As[0][0];
    const unsigned ldsA1 = (unsigned)(uintptr_t)&As[1][0];
    const unsigned ldsB0 = (unsigned)(uintptr_t)&Bs[0][0];
    const unsigned ldsB1 = (unsigned)(uintptr_t)&Bs[1][0];

    v8f acc[4][4];
#pragma unroll
    for (int mf = 0; mf < 4; ++mf)
#pragma unroll
        for (int nf = 0; nf < 4; ++nf) acc[mf][nf] = zero8();

    // prologue: DMA first K-tile into buffer 0
    if (wave == 0) {
        tdm_load_tile(gA, ldsA0, BM, K);
        tdm_load_tile(gB, ldsB0, BN, K);
    }

    int buf = 0;
    for (int k0 = 0; k0 < K; k0 += BK) {
        const bool pend = (k0 + BK < K);
        if (wave == 0) {
            if (pend) {   // DMA next tile into the other buffer (overlaps compute)
                tdm_load_tile(gA + (k0 + BK), buf ? ldsA0 : ldsA1, BM, K);
                tdm_load_tile(gB + (k0 + BK), buf ? ldsB0 : ldsB1, BN, K);
                __builtin_amdgcn_s_wait_tensorcnt(2);   // current pair complete
            } else {
                __builtin_amdgcn_s_wait_tensorcnt(0);
            }
        }
        __syncthreads();   // buf tile visible to all waves

        const __bf16* curA = &As[buf][0];
        const __bf16* curB = &Bs[buf][0];

        Frag af[4], bfr[4];
#pragma unroll
        for (int mf = 0; mf < 4; ++mf) {
            const __bf16* p = curA + (waveM * 64 + mf * 16 + lr) * LSTR;
            af[mf].q[0] = *(const uint4*)(p + lg * 16);       // K 0..7 / 16..23
            af[mf].q[1] = *(const uint4*)(p + lg * 16 + 8);   // K 8..15 / 24..31
        }
#pragma unroll
        for (int nf = 0; nf < 4; ++nf) {
            const __bf16* p = curB + (waveN * 64 + nf * 16 + lr) * LSTR + lg * 16;
            bfr[nf].q[0] = *(const uint4*)(p);
            bfr[nf].q[1] = *(const uint4*)(p + 8);
        }
#pragma unroll
        for (int mf = 0; mf < 4; ++mf)
#pragma unroll
            for (int nf = 0; nf < 4; ++nf)
                acc[mf][nf] = __builtin_amdgcn_wmma_f32_16x16x32_bf16(
                    false, af[mf].v, false, bfr[nf].v,
                    (short)0, acc[mf][nf], false, false);

        __syncthreads();   // buf free for the DMA two iterations ahead
        buf ^= 1;
    }

    // epilogue: bias + f32 store (lanes contiguous in N -> coalesced)
#pragma unroll
    for (int nf = 0; nf < 4; ++nf) {
        const int col = blockN + waveN * 64 + nf * 16 + lr;
        const float bv = bias[col];
#pragma unroll
        for (int mf = 0; mf < 4; ++mf) {
#pragma unroll
            for (int r = 0; r < 8; ++r) {
                const int row = blockM + waveM * 64 + mf * 16 + r + 8 * lg;
                C[(size_t)row * N + col] = acc[mf][nf][r] + bv;
            }
        }
    }
}

// ---------------------------------------------------------------------------
// Kernel 3: RMSNorm(full 2048 dim) + RoPE + pack to bf16 (B,H,S,HD)
// one block per (s, b, sel) row; 256 threads x 8 contiguous dims
// ---------------------------------------------------------------------------
__global__ __launch_bounds__(256) void normrope_kernel(
    const float* __restrict__ Qf, const float* __restrict__ Kf,
    const float* __restrict__ nqw, const float* __restrict__ nkw,
    const float* __restrict__ fcos, const float* __restrict__ fsin,
    __bf16* __restrict__ Qh, __bf16* __restrict__ Kh)
{
    const int s = blockIdx.x, b = blockIdx.y, sel = blockIdx.z;
    const float* src = (sel ? Kf : Qf) + ((size_t)b * S_ + s) * DIM_;
    const float* w   = sel ? nkw : nqw;
    __bf16*      dst = sel ? Kh : Qh;

    const int t  = threadIdx.x;
    const int c0 = t * 8;

    float x[8];
    float4 v0 = *(const float4*)(src + c0);
    float4 v1 = *(const float4*)(src + c0 + 4);
    x[0] = v0.x; x[1] = v0.y; x[2] = v0.z; x[3] = v0.w;
    x[4] = v1.x; x[5] = v1.y; x[6] = v1.z; x[7] = v1.w;

    float ss = 0.0f;
#pragma unroll
    for (int j = 0; j < 8; ++j) ss += x[j] * x[j];

    __shared__ float red[256];
    red[t] = ss;
    __syncthreads();
    for (int off = 128; off > 0; off >>= 1) {
        if (t < off) red[t] += red[t + off];
        __syncthreads();
    }
    const float rinv = rsqrtf(red[0] * (1.0f / (float)DIM_) + 1e-6f);

    float y[8];
#pragma unroll
    for (int j = 0; j < 8; ++j) y[j] = w[c0 + j] * x[j] * rinv;

    // RoPE: pair (2p, 2p+1) uses cos = fcos[s][2p], sin = fsin[s][2p+1]
    const int dh = c0 & (HD_ - 1);                // even, within-head offset
    const float* fc = fcos + (size_t)s * HD_;
    const float* fs = fsin + (size_t)s * HD_;
    Pack8 o;
#pragma unroll
    for (int j = 0; j < 8; j += 2) {
        const float cs = fc[dh + j];
        const float sn = fs[dh + j + 1];
        o.e[j]     = (__bf16)(y[j] * cs - y[j + 1] * sn);
        o.e[j + 1] = (__bf16)(y[j] * sn + y[j + 1] * cs);
    }
    const int h = c0 >> 7;
    const size_t di = (((size_t)b * H_ + h) * S_ + s) * HD_ + dh;
    *(uint4*)(dst + di) = o.u;
}

// ---------------------------------------------------------------------------
// Kernel 4: V (B,S,DIM) f32  ->  V^T (B,H,HD,S) bf16
// ---------------------------------------------------------------------------
__global__ __launch_bounds__(256) void vtrans_kernel(const float* __restrict__ Vf,
                                                     __bf16* __restrict__ Vt) {
    const int tid = blockIdx.x * 256 + threadIdx.x;
    const int s8 = tid & 255;                 // S_/8 = 256 groups
    const int c  = (tid >> 8) & (DIM_ - 1);
    const int b  = tid >> 19;                 // 256*2048 = 2^19 per batch
    if (b >= B_) return;
    const int s0 = s8 * 8;
    Pack8 o;
#pragma unroll
    for (int i = 0; i < 8; ++i)
        o.e[i] = (__bf16)Vf[((size_t)b * S_ + s0 + i) * DIM_ + c];
    const size_t di = (((size_t)b * H_ + (c >> 7)) * HD_ + (c & 127)) * (size_t)S_ + s0;
    *(uint4*)(Vt + di) = o.u;
}

// ---------------------------------------------------------------------------
// Kernel 5: flash attention, one wave per 16-query tile of one (b,h).
// S^T = K_tile * Q^T so the softmax key-axis lives in VGPRs per lane; online
// softmax needs only shfl_xor(16) pair-combines.  P is re-laid into a WMMA-B
// fragment with the same half-wave exchange, then out^T += V^T * P^T.
// ---------------------------------------------------------------------------
__global__ __launch_bounds__(128) void attn_kernel(
    const __bf16* __restrict__ Qh,   // (B,H,S,HD)
    const __bf16* __restrict__ Kh,   // (B,H,S,HD)
    const __bf16* __restrict__ Vt,   // (B,H,HD,S)
    __bf16* __restrict__ Ao)         // (B,S,DIM)
{
    const int wave = threadIdx.x >> 5;
    const int lane = threadIdx.x & 31;
    const int lg   = lane >> 4;
    const int lr   = lane & 15;

    const int qt = blockIdx.x * 4 + wave;    // query tile 0..127
    const int h  = blockIdx.y;
    const int b  = blockIdx.z;
    const int q0 = qt * 16;

    const __bf16* qbase = Qh + ((size_t)b * H_ + h) * (size_t)S_ * HD_;
    const __bf16* kbase = Kh + ((size_t)b * H_ + h) * (size_t)S_ * HD_;
    const __bf16* vbase = Vt + ((size_t)b * H_ + h) * (size_t)HD_ * S_;

    // Q tile as 4 B-fragments (K = HD split into 4x32), loaded once
    Frag qf[4];
#pragma unroll
    for (int df = 0; df < 4; ++df) {
        const __bf16* p = qbase + (size_t)(q0 + lr) * HD_ + df * 32;
        qf[df].q[0] = *(const uint4*)(p + lg * 8);
        qf[df].q[1] = *(const uint4*)(p + 16 + lg * 8);
    }

    v8f o[8];
#pragma unroll
    for (int dc = 0; dc < 8; ++dc) o[dc] = zero8();
    float mrun = -1e30f, lrun = 0.0f;
    const float scale = 0.08838834764831845f;   // 1/sqrt(128)

    for (int kb = 0; kb < S_; kb += 32) {
        // scores^T for two 16-key tiles
        v8f st0 = zero8(), st1 = zero8();
#pragma unroll
        for (int df = 0; df < 4; ++df) {
            Frag ka;
            const __bf16* p = kbase + (size_t)(kb + lr) * HD_ + df * 32;
            ka.q[0] = *(const uint4*)(p + lg * 8);
            ka.q[1] = *(const uint4*)(p + 16 + lg * 8);
            st0 = __builtin_amdgcn_wmma_f32_16x16x32_bf16(false, ka.v, false, qf[df].v,
                                                          (short)0, st0, false, false);
            Frag kc;
            p = kbase + (size_t)(kb + 16 + lr) * HD_ + df * 32;
            kc.q[0] = *(const uint4*)(p + lg * 8);
            kc.q[1] = *(const uint4*)(p + 16 + lg * 8);
            st1 = __builtin_amdgcn_wmma_f32_16x16x32_bf16(false, kc.v, false, qf[df].v,
                                                          (short)0, st1, false, false);
        }

        // per-lane: 16 key scores for query column lr; pair (lr,lg) <-> (lr,lg^1)
        float sA[8], sB[8];
        float bm = -1e30f;
#pragma unroll
        for (int r = 0; r < 8; ++r) {
            sA[r] = st0[r] * scale;            // key = kb + 8*lg + r
            sB[r] = st1[r] * scale;            // key = kb + 16 + 8*lg + r
            bm = fmaxf(bm, fmaxf(sA[r], sB[r]));
        }
        bm = fmaxf(bm, __shfl_xor(bm, 16, 32));
        const float mnew = fmaxf(mrun, bm);
        const float corr = __expf(mrun - mnew);
        float bs = 0.0f;
#pragma unroll
        for (int r = 0; r < 8; ++r) {
            sA[r] = __expf(sA[r] - mnew);
            sB[r] = __expf(sB[r] - mnew);
            bs += sA[r] + sB[r];
        }
        bs += __shfl_xor(bs, 16, 32);
        lrun = lrun * corr + bs;
        mrun = mnew;
#pragma unroll
        for (int dc = 0; dc < 8; ++dc)
#pragma unroll
            for (int e = 0; e < 8; ++e) o[dc][e] *= corr;

        // build P^T B-fragment (K = 32 keys of this block, col = query)
        float sAx[8], sBx[8];
#pragma unroll
        for (int r = 0; r < 8; ++r) {
            sAx[r] = __shfl_xor(sA[r], 16, 32);
            sBx[r] = __shfl_xor(sB[r], 16, 32);
        }
        Frag pf;
#pragma unroll
        for (int j = 0; j < 8; ++j) {
            pf.e[j]     = (__bf16)(lg ? sBx[j] : sA[j]);   // keys 0..7  / 16..23
            pf.e[8 + j] = (__bf16)(lg ? sB[j]  : sAx[j]);  // keys 8..15 / 24..31
        }

        // out^T += V^T * P^T   (8 d-chunks of 16)
#pragma unroll
        for (int dc = 0; dc < 8; ++dc) {
            Frag va;
            const __bf16* p = vbase + (size_t)(dc * 16 + lr) * S_ + kb;
            va.q[0] = *(const uint4*)(p + lg * 16);
            va.q[1] = *(const uint4*)(p + lg * 16 + 8);
            o[dc] = __builtin_amdgcn_wmma_f32_16x16x32_bf16(false, va.v, false, pf.v,
                                                            (short)0, o[dc], false, false);
        }
    }

    // normalize & store: lane's query = q0+lr, d = dc*16 + 8*lg + r (contiguous)
    const float inv = 1.0f / lrun;
#pragma unroll
    for (int dc = 0; dc < 8; ++dc) {
        Pack8 ob;
#pragma unroll
        for (int r = 0; r < 8; ++r) ob.e[r] = (__bf16)(o[dc][r] * inv);
        const size_t di = ((size_t)b * S_ + q0 + lr) * DIM_ + h * HD_ + dc * 16 + lg * 8;
        *(uint4*)(Ao + di) = ob.u;
    }
}

// ---------------------------------------------------------------------------
// Host-side orchestration
// ---------------------------------------------------------------------------
extern "C" void kernel_launch(void* const* d_in, const int* in_sizes, int n_in,
                              void* d_out, int out_size, void* d_ws, size_t ws_size,
                              hipStream_t stream) {
    const float* hidden = (const float*)d_in[0];
    const float* fcos   = (const float*)d_in[1];
    const float* fsin   = (const float*)d_in[2];
    const float* Wq     = (const float*)d_in[3];
    const float* bq     = (const float*)d_in[4];
    const float* Wk     = (const float*)d_in[5];
    const float* bk     = (const float*)d_in[6];
    const float* Wv     = (const float*)d_in[7];
    const float* bv     = (const float*)d_in[8];
    const float* nqw    = (const float*)d_in[9];
    const float* nkw    = (const float*)d_in[10];
    const float* Wo     = (const float*)d_in[11];
    const float* bo     = (const float*)d_in[12];
    float* out = (float*)d_out;

    char* ws = (char*)d_ws;
    size_t off = 0;
    auto alloc = [&](size_t bytes) -> void* {
        void* p = ws + off;
        off += (bytes + 255) & ~(size_t)255;
        return p;
    };

    const size_t NTOK = (size_t)MROWS * DIM_;    // 8,388,608 elements
    const size_t NW   = (size_t)DIM_ * DIM_;     // 4,194,304 elements

    __bf16* Xb   = (__bf16*)alloc(NTOK * 2);
    __bf16* Wqkv = (__bf16*)alloc(3 * NW * 2);   // Wq|Wk|Wv stacked
    __bf16* Wob  = (__bf16*)alloc(NW * 2);
    float*  Qf   = (float*)alloc(NTOK * 4);
    float*  Kf   = (float*)alloc(NTOK * 4);
    float*  Vf   = (float*)alloc(NTOK * 4);
    __bf16* Qh   = (__bf16*)alloc(NTOK * 2);     // (B,H,S,HD)
    __bf16* Kh   = (__bf16*)alloc(NTOK * 2);     // (B,H,S,HD)
    __bf16* Vt   = (__bf16*)alloc(NTOK * 2);     // (B,H,HD,S)
    __bf16* Ao   = (__bf16*)alloc(NTOK * 2);     // (B,S,DIM)

    // 1) f32 -> bf16 conversions
    cvt_f32_bf16<<<(int)(NTOK / 1024), 256, 0, stream>>>(hidden, Xb, (int)NTOK);
    cvt_f32_bf16<<<(int)(NW / 1024), 256, 0, stream>>>(Wq, Wqkv + 0 * NW, (int)NW);
    cvt_f32_bf16<<<(int)(NW / 1024), 256, 0, stream>>>(Wk, Wqkv + 1 * NW, (int)NW);
    cvt_f32_bf16<<<(int)(NW / 1024), 256, 0, stream>>>(Wv, Wqkv + 2 * NW, (int)NW);
    cvt_f32_bf16<<<(int)(NW / 1024), 256, 0, stream>>>(Wo, Wob, (int)NW);

    // 2) fused QKV projections: grid.z selects q/k/v
    {
        dim3 grid(DIM_ / BN, MROWS / BM, 3);
        gemm_nt_bias<<<grid, 256, 0, stream>>>(Xb, Wqkv, bq, bk, bv,
                                               Qf, Kf, Vf, MROWS, DIM_, DIM_);
    }

    // 3) RMSNorm + RoPE for Q and K
    {
        dim3 grid(S_, B_, 2);
        normrope_kernel<<<grid, 256, 0, stream>>>(Qf, Kf, nqw, nkw, fcos, fsin, Qh, Kh);
    }

    // 4) V -> V^T bf16
    vtrans_kernel<<<(B_ * DIM_ * (S_ / 8)) / 256, 256, 0, stream>>>(Vf, Vt);

    // 5) attention
    {
        dim3 grid(S_ / 64, H_, B_);   // 4 waves/block, 16 queries/wave
        attn_kernel<<<grid, 128, 0, stream>>>(Qh, Kh, Vt, Ao);
    }

    // 6) output projection -> d_out (f32)
    {
        dim3 grid(DIM_ / BN, MROWS / BM, 1);
        gemm_nt_bias<<<grid, 256, 0, stream>>>(Ao, Wob, bo, bo, bo,
                                               out, out, out, MROWS, DIM_, DIM_);
    }
}
